// VectorQuantizer_53566832115832
// MI455X (gfx1250) — matrix operand order repaired and checked by reading, not compiled, not run on hardware
//
#include <hip/hip_runtime.h>
#include <hip/hip_bf16.h>

// Problem constants (match reference)
#define NUM_EMB 8192
#define DIM     512
#define BATCH_N 32768
#define CCOST   0.25f

#define BM      128                 // batch rows per block (8 waves x 16)
#define NTILES  (NUM_EMB / 16)      // 512 codebook tiles of 16 codes
#define KCHUNKS (DIM / 32)          // 16 K-chunks of 32
#define SBS     528                 // padded halves per staged code row (512+16)

typedef __attribute__((ext_vector_type(16))) _Float16 v16h;
typedef __attribute__((ext_vector_type(8)))  float    v8f;

// ---------------------------------------------------------------------------
// Kernel 1: per-code squared norms; also zero counts and the SSE accumulator.
// ---------------------------------------------------------------------------
__global__ __launch_bounds__(128)
void vq_norms_kernel(const float* __restrict__ codebook,
                     float* __restrict__ norms,
                     float* __restrict__ counts,
                     float* __restrict__ sse) {
  const int row = blockIdx.x;
  const int tid = threadIdx.x;                       // 128 threads, DIM/4 float4s
  const float4* cb4 = (const float4*)(codebook + (size_t)row * DIM);
  float4 v = cb4[tid];
  float s = v.x * v.x + v.y * v.y + v.z * v.z + v.w * v.w;
  #pragma unroll
  for (int m = 16; m >= 1; m >>= 1) s += __shfl_xor(s, m, 32);
  __shared__ float red[4];
  if ((tid & 31) == 0) red[tid >> 5] = s;
  __syncthreads();
  if (tid == 0) {
    norms[row]  = red[0] + red[1] + red[2] + red[3];
    counts[row] = 0.0f;
    if (row == 0) *sse = 0.0f;
  }
}

// ---------------------------------------------------------------------------
// Kernel 2: WMMA distance-argmax kernel.
//   score(b,n) = x_b . e_n - 0.5*||e_n||^2   (maximize  <=>  minimize dist)
// Block: 256 threads = 8 waves; wave w owns rows [blk*128 + 16w, +16).
// A fragments (f16) for full K held in registers; B tiles double-buffered in
// LDS with cooperative f32->f16 staging.
// ---------------------------------------------------------------------------
__global__ __launch_bounds__(256)
void vq_argmin_kernel(const float* __restrict__ inputs,
                      const float* __restrict__ codebook,
                      const float* __restrict__ norms,
                      int* __restrict__ indices) {
  __shared__ __align__(16) _Float16 sB[2][16 * SBS];   // 2 x 16.5KB

  const int tid  = threadIdx.x;
  const int wave = tid >> 5;
  const int lane = tid & 31;
  const int l16  = lane & 15;
  const int hi   = lane >> 4;                          // 0 or 1 (lane half)

  const int row0 = blockIdx.x * BM + wave * 16;

  // ---- Load A fragments once: ISA 16-bit A layout, converting f32 -> f16.
  // Lane holds row (row0 + l16); halves 0..7 = K kb..kb+7, halves 8..15 =
  // K kb+16..kb+23, with kb = c*32 + hi*8.
  v16h afrag[KCHUNKS];
  {
    const float* arow = inputs + (size_t)(row0 + l16) * DIM;
    #pragma unroll
    for (int c = 0; c < KCHUNKS; ++c) {
      const int kb = c * 32 + hi * 8;
      const float4 f0 = *(const float4*)(arow + kb);
      const float4 f1 = *(const float4*)(arow + kb + 4);
      const float4 f2 = *(const float4*)(arow + kb + 16);
      const float4 f3 = *(const float4*)(arow + kb + 20);
      v16h a;
      a[0]  = (_Float16)f0.x; a[1]  = (_Float16)f0.y;
      a[2]  = (_Float16)f0.z; a[3]  = (_Float16)f0.w;
      a[4]  = (_Float16)f1.x; a[5]  = (_Float16)f1.y;
      a[6]  = (_Float16)f1.z; a[7]  = (_Float16)f1.w;
      a[8]  = (_Float16)f2.x; a[9]  = (_Float16)f2.y;
      a[10] = (_Float16)f2.z; a[11] = (_Float16)f2.w;
      a[12] = (_Float16)f3.x; a[13] = (_Float16)f3.y;
      a[14] = (_Float16)f3.z; a[15] = (_Float16)f3.w;
      afrag[c] = a;
    }
  }

  float best_s[8];
  int   best_i[8];
  #pragma unroll
  for (int m = 0; m < 8; ++m) { best_s[m] = -3.402823466e38f; best_i[m] = 0; }

  // Cooperative stage of one 16-code tile into LDS buffer `buf` (f32 -> f16).
  auto stage = [&](int buf, int t) {
    const int n0 = t * 16;
    #pragma unroll
    for (int j = 0; j < 8; ++j) {
      const int e4   = tid + j * 256;      // float4 index within 16x512 tile
      const int base = e4 * 4;
      const int ci   = base >> 9;          // code within tile
      const int k    = base & 511;
      const float4 f = *(const float4*)(codebook + (size_t)(n0 + ci) * DIM + k);
      _Float16* d = &sB[buf][ci * SBS + k];
      d[0] = (_Float16)f.x; d[1] = (_Float16)f.y;
      d[2] = (_Float16)f.z; d[3] = (_Float16)f.w;
    }
  };

  stage(0, 0);
  __syncthreads();

  int cur = 0;
  for (int t = 0; t < NTILES; ++t) {
    if (t + 1 < NTILES) stage(cur ^ 1, t + 1);

    // B fragments: column = l16, halves 0..7 = K c*32 + hi*8 .. +7, halves 8..15 = K c*32 + 16 + hi*8 .. +7.
    const _Float16* bbase = &sB[cur][l16 * SBS + hi * 8];
    v8f acc = {};
    #pragma unroll
    for (int c = 0; c < KCHUNKS; ++c) {
      const _Float16* bp = bbase + c * 32;
      v16h b;
      #pragma unroll
      for (int h = 0; h < 16; ++h) b[h] = bp[(h & 7) + ((h >> 3) << 4)];
      acc = __builtin_amdgcn_wmma_f32_16x16x32_f16(
          /*neg_a=*/false, afrag[c], /*neg_b=*/false, b,
          /*c_mod=*/(short)0, acc, /*reuse_a=*/false, /*reuse_b=*/false);
    }

    const int   n  = t * 16 + l16;
    const float sc = -0.5f * norms[n];
    #pragma unroll
    for (int m = 0; m < 8; ++m) {
      const float s = acc[m] + sc;
      if (s > best_s[m]) { best_s[m] = s; best_i[m] = n; }
    }

    __syncthreads();
    cur ^= 1;
  }

  // Butterfly argmax across the 16 lanes holding the same row
  // (first-index tie-break to match jnp.argmin).
  #pragma unroll
  for (int m = 0; m < 8; ++m) {
    float s = best_s[m];
    int   i = best_i[m];
    #pragma unroll
    for (int msk = 8; msk >= 1; msk >>= 1) {
      const float os = __shfl_xor(s, msk, 32);
      const int   oi = __shfl_xor(i, msk, 32);
      if (os > s || (os == s && oi < i)) { s = os; i = oi; }
    }
    best_s[m] = s; best_i[m] = i;
  }
  if (l16 == 0) {
    const int rbase = row0 + hi * 8;   // lanes<16 -> rows M=0..7, lanes>=16 -> 8..15
    #pragma unroll
    for (int m = 0; m < 8; ++m) indices[rbase + m] = best_i[m];
  }
}

// ---------------------------------------------------------------------------
// Kernel 3: gather quantized rows, straight-through output, SSE + counts.
// ---------------------------------------------------------------------------
__global__ __launch_bounds__(128)
void vq_gather_kernel(const float* __restrict__ inputs,
                      const float* __restrict__ codebook,
                      const int* __restrict__ indices,
                      float* __restrict__ out,
                      float* __restrict__ counts,
                      float* __restrict__ sse) {
  const int row = blockIdx.x;
  const int tid = threadIdx.x;                 // 128 threads x float4 = 512
  const int idx = indices[row];

  const float4 x = ((const float4*)(inputs   + (size_t)row * DIM))[tid];
  const float4 q = ((const float4*)(codebook + (size_t)idx * DIM))[tid];
  float4 d, o;
  d.x = q.x - x.x; d.y = q.y - x.y; d.z = q.z - x.z; d.w = q.w - x.w;
  o.x = x.x + d.x; o.y = x.y + d.y; o.z = x.z + d.z; o.w = x.w + d.w;
  ((float4*)(out + (size_t)row * DIM))[tid] = o;

  float s = d.x * d.x + d.y * d.y + d.z * d.z + d.w * d.w;
  #pragma unroll
  for (int m = 16; m >= 1; m >>= 1) s += __shfl_xor(s, m, 32);
  __shared__ float red[4];
  if ((tid & 31) == 0) red[tid >> 5] = s;
  __syncthreads();
  if (tid == 0) {
    atomicAdd(sse, red[0] + red[1] + red[2] + red[3]);
    atomicAdd(&counts[idx], 1.0f);
  }
}

// ---------------------------------------------------------------------------
// Kernel 4: loss and perplexity scalars.
// ---------------------------------------------------------------------------
__global__ __launch_bounds__(256)
void vq_finalize_kernel(const float* __restrict__ counts,
                        const float* __restrict__ sse,
                        float* __restrict__ out_scalars) {
  const int tid = threadIdx.x;
  float h = 0.0f;
  for (int i = tid; i < NUM_EMB; i += 256) {
    const float p = counts[i] * (1.0f / (float)BATCH_N);
    h += p * logf(p + 1e-10f);
  }
  #pragma unroll
  for (int m = 16; m >= 1; m >>= 1) h += __shfl_xor(h, m, 32);
  __shared__ float red[8];
  if ((tid & 31) == 0) red[tid >> 5] = h;
  __syncthreads();
  if (tid == 0) {
    float t = 0.0f;
    #pragma unroll
    for (int i = 0; i < 8; ++i) t += red[i];
    const float mse = (*sse) / ((float)BATCH_N * (float)DIM);
    out_scalars[0] = mse + CCOST * mse;   // q_latent_loss + c * e_latent_loss
    out_scalars[1] = expf(-t);            // perplexity
  }
}

// ---------------------------------------------------------------------------
extern "C" void kernel_launch(void* const* d_in, const int* in_sizes, int n_in,
                              void* d_out, int out_size, void* d_ws, size_t ws_size,
                              hipStream_t stream) {
  const float* inputs   = (const float*)d_in[0];   // [32768, 512] f32
  const float* codebook = (const float*)d_in[1];   // [8192, 512] f32
  float* out = (float*)d_out;                      // 32768*512 + 2 f32

  // Workspace layout (~196KB):
  char*  ws      = (char*)d_ws;
  float* norms   = (float*)ws;                                   // 8192 f32
  int*   indices = (int*)(ws + NUM_EMB * 4);                     // 32768 i32
  float* counts  = (float*)(ws + NUM_EMB * 4 + BATCH_N * 4);     // 8192 f32
  float* sse     = (float*)(ws + NUM_EMB * 8 + BATCH_N * 4);     // 1 f32

  vq_norms_kernel<<<NUM_EMB, 128, 0, stream>>>(codebook, norms, counts, sse);
  vq_argmin_kernel<<<BATCH_N / BM, 256, 0, stream>>>(inputs, codebook, norms, indices);
  vq_gather_kernel<<<BATCH_N, 128, 0, stream>>>(inputs, codebook, indices, out, counts, sse);
  vq_finalize_kernel<<<1, 256, 0, stream>>>(counts, sse, out + (size_t)BATCH_N * DIM);
}
